// LSTMModel1_43911745634896
// MI455X (gfx1250) — matrix operand order repaired and checked by reading, not compiled, not run on hardware
//
#include <hip/hip_runtime.h>

#if defined(__has_include)
#  if __has_include(<hip/amd_detail/amd_gfx1250_TDM.h>)
#    define ATH_TDM_6ARG 1
#  endif
#endif

typedef __attribute__((ext_vector_type(16))) __bf16   v16bf;
typedef __attribute__((ext_vector_type(8)))  float    v8f;
typedef __attribute__((ext_vector_type(4)))  unsigned v4u;
typedef __attribute__((ext_vector_type(8)))  int      v8i;
typedef __attribute__((ext_vector_type(4)))  int      v4i;

#define K_B    32
#define K_T    1024
#define K_IN   64
#define K_H1   512
#define K_G    2048      // 4*H1
#define K_H2   256
#define K_MTOT 32768     // B*T
#define K_NWG  16

// ---------------------------------------------------------------- helpers
static __device__ __forceinline__ __bf16 f2bf(float f) {
  union { float f; unsigned u; } v; v.f = f;
  unsigned r = (v.u + 0x7FFFu + ((v.u >> 16) & 1u)) >> 16;
  unsigned short s = (unsigned short)r;
  return __builtin_bit_cast(__bf16, s);
}
static __device__ __forceinline__ float sigmoidf(float x) {
  return 1.f / (1.f + __expf(-x));
}
static __device__ __forceinline__ v8f wmma_bf16(v16bf a, v16bf b, v8f c) {
  return __builtin_amdgcn_wmma_f32_16x16x32_bf16(false, a, false, b, (short)0, c,
                                                 false, false);
}

// TDM: fetch a 3D tile (x=32 floats, y=4 gates stride 512, z=32 rows stride T*G)
// from the precomputed gate buffer into LDS. D# layout per CDNA5 ISA ch.8.
#if __has_builtin(__builtin_amdgcn_tensor_load_to_lds)
#define ATH_HAVE_TDM 1
static __device__ __forceinline__ void tdm_fetch_xg(const float* gaddr,
                                                    unsigned lds_off) {
  unsigned long long ga = (unsigned long long)(size_t)gaddr;
  v4u g0;
  g0[0] = 1u;                                   // count=1, user descriptor
  g0[1] = lds_off;                              // lds_addr
  g0[2] = (unsigned)(ga & 0xFFFFFFFFull);       // global_addr[31:0]
  g0[3] = (unsigned)((ga >> 32) & 0x01FFFFFFull) | (2u << 30); // addr hi | type=2
  v8i g1;
  g1[0] = (int)(2u << 16);                      // data_size = 4 bytes
  g1[1] = (int)(32u << 16);                     // tensor_dim0 lo16 @ [31:16]
  g1[2] = (int)(4u << 16);                      // td0 hi16 | tensor_dim1 lo16
  g1[3] = (int)(32u << 16);                     // td1 hi16 | tile_dim0 = 32
  g1[4] = (int)(4u | (32u << 16));              // tile_dim1=4 | tile_dim2=32
  g1[5] = 512;                                  // tensor_dim0_stride lo32 (=512)
  g1[6] = 0;                                    // stride0 hi16 | stride1 lo16
  g1[7] = (int)(2097152u >> 16);                // stride1 (=T*G) bits 47:16
  v4i g2; g2[0] = 32; g2[1] = 0; g2[2] = 0; g2[3] = 0;  // tensor_dim2=32
  v4i g3 = (v4i){0, 0, 0, 0};
#if defined(ATH_TDM_6ARG)
  v8i gx = (v8i){0, 0, 0, 0, 0, 0, 0, 0};
  __builtin_amdgcn_tensor_load_to_lds(g0, g1, g2, g3, gx, 0);
#else
  __builtin_amdgcn_tensor_load_to_lds(g0, g1, g2, g3, 0);
#endif
}
#endif

// ---------------------------------------------------------------- packing
// B-fragment layout [nt][kt][lane(32)][j(16)]: lane -> col n=nt*16+lane%16,
// K = kt*32 + (lane/16)*8 + (j%8) + (j/8)*16.   W is row-major (N,K); B[k][n] = W[n][k].
__global__ void pack_b_kernel(const float* __restrict__ W, __bf16* __restrict__ dst,
                              int N, int K) {
  int e = blockIdx.x * blockDim.x + threadIdx.x;
  if (e >= N * K) return;
  int KT = K >> 5;
  int frag = e >> 9, lane = (e >> 4) & 31, j = e & 15;
  int nt = frag / KT, kt = frag % KT;
  int n = nt * 16 + (lane & 15);
  int k = kt * 32 + ((lane >> 4) << 3) + (j & 7) + ((j >> 3) << 4);
  dst[e] = f2bf(W[(size_t)n * K + k]);
}

// A-fragment layout [rt][kt][lane(32)][j(16)]: lane -> row m=rt*16+lane%16,
// K = kt*32 + (lane/16)*8 + (j%8) + (j/8)*16.  Rows = MTOT, A row-major.
__global__ void pack_a_kernel(const float* __restrict__ A, __bf16* __restrict__ dst,
                              int K) {
  int e = blockIdx.x * blockDim.x + threadIdx.x;
  if (e >= K_MTOT * K) return;
  int KT = K >> 5;
  int frag = e >> 9, lane = (e >> 4) & 31, j = e & 15;
  int rt = frag / KT, kt = frag % KT;
  int m = rt * 16 + (lane & 15);
  int k = kt * 32 + ((lane >> 4) << 3) + (j & 7) + ((j >> 3) << 4);
  dst[e] = f2bf(A[(size_t)m * K + k]);
}

__global__ void zero_bf16_kernel(__bf16* p, int n) {
  int i = blockIdx.x * blockDim.x + threadIdx.x;
  if (i < n) p[i] = f2bf(0.f);
}
__global__ void zero_u32_kernel(unsigned* p, int n) {
  int i = blockIdx.x * blockDim.x + threadIdx.x;
  if (i < n) p[i] = 0u;
}

// ---------------------------------------------------------------- big GEMM
// C(m,n) = sum_k A(m,k)*B(k,n) + bias0(n) [+ bias1(n)], optional ReLU.
// Register-blocked: each wave computes a 64x64 output block (4x4 WMMA tiles),
// 16 WMMAs per 8 fragment loads -> 4x fragment reuse vs 1 tile/wave.
__global__ __launch_bounds__(256) void wmma_gemm_bias(
    const __bf16* __restrict__ Apack, const __bf16* __restrict__ Bpack,
    const float* __restrict__ bias0, const float* __restrict__ bias1,
    float* __restrict__ C, int NB, int KT, int N, int relu) {
  const int lane = threadIdx.x & 31;
  const int wave = threadIdx.x >> 5;
  const int blk = blockIdx.x * 8 + wave;
  const int rb = blk / NB, nb = blk % NB;     // 64-row block, 64-col block
  v8f acc[4][4] = {};
  const __bf16* abase = Apack + ((size_t)(rb * 4) * KT) * 512 + lane * 16;
  const __bf16* bbase = Bpack + ((size_t)(nb * 4) * KT) * 512 + lane * 16;
  for (int k = 0; k < KT; ++k) {
    v16bf a[4], b[4];
#pragma unroll
    for (int i = 0; i < 4; ++i) {
      a[i] = *reinterpret_cast<const v16bf*>(abase + ((size_t)i * KT + k) * 512);
      b[i] = *reinterpret_cast<const v16bf*>(bbase + ((size_t)i * KT + k) * 512);
    }
#pragma unroll
    for (int i = 0; i < 4; ++i)
#pragma unroll
      for (int j = 0; j < 4; ++j)
        acc[i][j] = wmma_bf16(a[i], b[j], acc[i][j]);
  }
  const int ncol0 = nb * 64 + (lane & 15);
  const int mrow0 = rb * 64 + ((lane >> 4) << 3);
#pragma unroll
  for (int j = 0; j < 4; ++j) {
    int n = ncol0 + j * 16;
    float bs = bias0[n];
    if (bias1) bs += bias1[n];
#pragma unroll
    for (int i = 0; i < 4; ++i) {
      int mb = mrow0 + i * 16;
#pragma unroll
      for (int v = 0; v < 8; ++v) {
        float r = acc[i][j][v] + bs;
        if (relu) r = fmaxf(r, 0.f);
        C[(size_t)(mb + v) * N + n] = r;
      }
    }
  }
}

// ---------------------------------------------------------------- recurrence
// Persistent 16-WG gang. WG `wg` owns h columns [wg*32, wg*32+32) and computes
// the i/f/g/o gate slices for them (M=32, N=128, K=512) per step with bf16
// WMMA; new h is written bf16 in A-fragment layout to a double-buffered global
// broadcast buffer (L2-resident). xg slice prefetched to LDS via TDM.
__global__ __launch_bounds__(256) void lstm_recurrent_kernel(
    const float* __restrict__ xg,        // (MTOT, G) rows m = b*T + t
    const __bf16* __restrict__ whh,      // B-pack [128 NT][16 KT][32][16]
    __bf16* __restrict__ hbuf,           // 2 x A-pack [2 mt][16 kt][32][16]
    float* __restrict__ hs, float* __restrict__ cs,  // (B,T,H1)
    unsigned* __restrict__ sync_cnt) {
  __shared__ float lds_gates[32][128];
  __shared__ float lds_c[32][32];
  __shared__ float lds_xg[32 * 4 * 32];
  const int wg = blockIdx.x;
  const int tid = threadIdx.x;
  const int lane = tid & 31, wave = tid >> 5;

  for (int p = tid; p < 32 * 32; p += 256) lds_c[p >> 5][p & 31] = 0.f;

  // wave tile assignment: 16 output tiles (2 mt x 8 local n-tiles), 2 per wave
  const int tile0 = wave * 2;
  const int mt = tile0 >> 3;
  const int n8a = tile0 & 7, n8b = n8a + 1;           // same gate pair
  const int NTa = (n8a >> 1) * 32 + wg * 2 + (n8a & 1);
  const int NTb = (n8b >> 1) * 32 + wg * 2 + (n8b & 1);
  const __bf16* bpa = whh + ((size_t)NTa * 16) * 512 + lane * 16;
  const __bf16* bpb = whh + ((size_t)NTb * 16) * 512 + lane * 16;
  const int nla = n8a * 16 + (lane & 15);
  const int nlb = n8b * 16 + (lane & 15);
  const int mbase = mt * 16 + ((lane >> 4) << 3);
#if defined(ATH_HAVE_TDM)
  const unsigned lds_xg_off = (unsigned)(size_t)(void*)&lds_xg[0];
#endif
  __syncthreads();

  for (int t = 0; t < K_T; ++t) {
#if defined(ATH_HAVE_TDM)
    if (wave == 0) tdm_fetch_xg(xg + (size_t)t * K_G + wg * 32, lds_xg_off);
#else
    for (int p = tid; p < 4096; p += 256) {
      int m = p >> 7, rem = p & 127;
      lds_xg[p] = xg[((size_t)m * K_T + t) * K_G + (rem >> 5) * 512 + wg * 32 +
                     (rem & 31)];
    }
#endif
    const __bf16* hr = hbuf + (size_t)(t & 1) * 16384;
    __bf16* hw = hbuf + (size_t)((t + 1) & 1) * 16384;

    v8f accA = {}, accB = {};
    const __bf16* ap = hr + ((size_t)mt * 16) * 512 + lane * 16;
#pragma unroll 4
    for (int kt = 0; kt < 16; ++kt) {
      v16bf a  = *reinterpret_cast<const v16bf*>(ap  + (size_t)kt * 512);
      v16bf ba = *reinterpret_cast<const v16bf*>(bpa + (size_t)kt * 512);
      v16bf bb = *reinterpret_cast<const v16bf*>(bpb + (size_t)kt * 512);
      accA = wmma_bf16(a, ba, accA);
      accB = wmma_bf16(a, bb, accB);
    }
#if defined(ATH_HAVE_TDM)
    __builtin_amdgcn_s_wait_tensorcnt(0);
#endif
    __syncthreads();  // lds_xg valid

    // gate pre-activations = WMMA acc + input contribution -> LDS staging
#pragma unroll
    for (int v = 0; v < 8; ++v) {
      int m = mbase + v;
      lds_gates[m][nla] = accA[v] + lds_xg[(m * 4 + (nla >> 5)) * 32 + (nla & 31)];
      lds_gates[m][nlb] = accB[v] + lds_xg[(m * 4 + (nlb >> 5)) * 32 + (nlb & 31)];
    }
    __syncthreads();

    // elementwise LSTM cell on owned 32-column slice (1024 elems / 256 thr)
    for (int p = tid; p < 1024; p += 256) {
      int m = p >> 5, hc = p & 31;
      float gi = sigmoidf(lds_gates[m][hc]);
      float gf = sigmoidf(lds_gates[m][32 + hc]);
      float gg = tanhf(lds_gates[m][64 + hc]);
      float go = sigmoidf(lds_gates[m][96 + hc]);
      float c = gf * lds_c[m][hc] + gi * gg;
      float h = go * tanhf(c);
      lds_c[m][hc] = c;
      size_t o = ((size_t)m * K_T + t) * K_H1 + wg * 32 + hc;
      hs[o] = h;
      cs[o] = c;
      // write h in WMMA A-fragment layout (kt == wg since slice is 32-aligned)
      int half = (hc >> 3) & 1;
      int lj = (hc & 7) + ((hc >> 4) << 3);
      int plane = (m & 15) + (half << 4);
      hw[(((m >> 4) * 16 + wg) * 32 + plane) * 16 + lj] = f2bf(h);
    }
    __threadfence();
    __syncthreads();

    // gang-wide sync: cluster barrier (NOP if not cluster-launched) + atomics
    if (tid == 0)
      __hip_atomic_fetch_add(sync_cnt, 1u, __ATOMIC_ACQ_REL,
                             __HIP_MEMORY_SCOPE_AGENT);
#if __has_builtin(__builtin_amdgcn_s_cluster_barrier)
    __builtin_amdgcn_s_cluster_barrier();
#endif
    if (tid == 0) {
      unsigned target = (unsigned)(K_NWG * (t + 1));
      while (__hip_atomic_load(sync_cnt, __ATOMIC_ACQUIRE,
                               __HIP_MEMORY_SCOPE_AGENT) < target)
        __builtin_amdgcn_s_sleep(1);
    }
    __syncthreads();
  }
}

// ---------------------------------------------------------------- head
__global__ void head_kernel(const float* __restrict__ hidden,
                            const float* __restrict__ W2,
                            const float* __restrict__ b2,
                            float* __restrict__ out) {
  int m = blockIdx.x * blockDim.x + threadIdx.x;
  if (m >= K_MTOT) return;
  float s = b2[0];
  const float* hp = hidden + (size_t)m * K_H2;
#pragma unroll 8
  for (int k = 0; k < K_H2; ++k) s += hp[k] * W2[k];
  out[m] = s;
}

// ---------------------------------------------------------------- launcher
extern "C" void kernel_launch(void* const* d_in, const int* in_sizes, int n_in,
                              void* d_out, int out_size, void* d_ws,
                              size_t ws_size, hipStream_t stream) {
  (void)in_sizes; (void)n_in; (void)out_size; (void)ws_size;
  const float* x    = (const float*)d_in[0];
  const float* Wih0 = (const float*)d_in[1];
  const float* Whh0 = (const float*)d_in[2];
  const float* bih0 = (const float*)d_in[3];
  const float* bhh0 = (const float*)d_in[4];
  const float* Wih1 = (const float*)d_in[5];
  const float* Whh1 = (const float*)d_in[6];
  const float* bih1 = (const float*)d_in[7];
  const float* bhh1 = (const float*)d_in[8];
  const float* W1   = (const float*)d_in[9];
  const float* b1   = (const float*)d_in[10];
  const float* W2   = (const float*)d_in[11];
  const float* b2   = (const float*)d_in[12];

  float* out = (float*)d_out;                 // (B,T,1)
  float* h0s = out + 32768;                   // (B,T,512)
  float* h1s = h0s + 16777216;
  float* c0s = h1s + 16777216;
  float* c1s = c0s + 16777216;

  char* ws = (char*)d_ws;
  float*  xg     = (float*)(ws);                          // 256 MB, reused
  __bf16* apack  = (__bf16*)(ws + 268435456ull);          // 32 MB
  __bf16* pWih0  = (__bf16*)(ws + 301989888ull);          // 256 KB
  __bf16* pWhh0  = (__bf16*)(ws + 302252032ull);          // 2 MB
  __bf16* pWih1  = (__bf16*)(ws + 304349184ull);          // 2 MB
  __bf16* pWhh1  = (__bf16*)(ws + 306446336ull);          // 2 MB
  __bf16* pW1    = (__bf16*)(ws + 308543488ull);          // 256 KB
  float*  hidden = (float*)(ws + 308805632ull);           // 32 MB
  __bf16* hbuf0  = (__bf16*)(ws + 342360064ull);          // 64 KB (2 bufs)
  __bf16* hbuf1  = (__bf16*)(ws + 342425600ull);          // 64 KB
  unsigned* sync = (unsigned*)(ws + 342491136ull);

  // state init (graph-replay safe: re-zeroed every call)
  zero_bf16_kernel<<<(65536 + 255) / 256, 256, 0, stream>>>(hbuf0, 65536);
  zero_u32_kernel<<<1, 32, 0, stream>>>(sync, 2);

  // weight packing (bf16, WMMA B-fragment order)
  pack_b_kernel<<<(2048 * 64 + 255) / 256, 256, 0, stream>>>(Wih0, pWih0, 2048, 64);
  pack_b_kernel<<<(2048 * 512 + 255) / 256, 256, 0, stream>>>(Whh0, pWhh0, 2048, 512);
  pack_b_kernel<<<(2048 * 512 + 255) / 256, 256, 0, stream>>>(Wih1, pWih1, 2048, 512);
  pack_b_kernel<<<(2048 * 512 + 255) / 256, 256, 0, stream>>>(Whh1, pWhh1, 2048, 512);
  pack_b_kernel<<<(256 * 512 + 255) / 256, 256, 0, stream>>>(W1, pW1, 256, 512);

  // ---- layer 0 ----  (xg GEMM: 512 row-blocks x 32 col-blocks of 64x64)
  pack_a_kernel<<<(K_MTOT * 64 + 255) / 256, 256, 0, stream>>>(x, apack, 64);
  wmma_gemm_bias<<<(512 * 32) / 8, 256, 0, stream>>>(apack, pWih0, bih0, bhh0,
                                                     xg, 32, 2, 2048, 0);
  lstm_recurrent_kernel<<<K_NWG, 256, 0, stream>>>(xg, pWhh0, hbuf0, h0s, c0s,
                                                   sync + 0);

  // ---- layer 1 ----
  pack_a_kernel<<<(K_MTOT * 512 + 255) / 256, 256, 0, stream>>>(h0s, apack, 512);
  wmma_gemm_bias<<<(512 * 32) / 8, 256, 0, stream>>>(apack, pWih1, bih1, bhh1,
                                                     xg, 32, 16, 2048, 0);
  lstm_recurrent_kernel<<<K_NWG, 256, 0, stream>>>(xg, pWhh1, hbuf1, h1s, c1s,
                                                   sync + 1);

  // ---- MLP head ----  (512 x 4 blocks of 64x64)
  pack_a_kernel<<<(K_MTOT * 512 + 255) / 256, 256, 0, stream>>>(h1s, apack, 512);
  wmma_gemm_bias<<<(512 * 4) / 8, 256, 0, stream>>>(apack, pW1, b1, nullptr,
                                                    hidden, 4, 16, 256, 1);
  head_kernel<<<(K_MTOT + 255) / 256, 256, 0, stream>>>(hidden, W2, b2, out);
}
